// DeepSeekMoE_27822798144110
// MI455X (gfx1250) — matrix operand order repaired and checked by reading, not compiled, not run on hardware
//
#include <hip/hip_runtime.h>
#include <hip/hip_bf16.h>
#include <math.h>

// ---------------------------------------------------------------------------
// DeepSeek-style MoE FFN for MI455X (gfx1250, wave32, WMMA).
// GEMMs: bf16 x bf16 -> f32 accum via v_wmma_f32_16x16x32_bf16.
// Routed experts: sparse top-2 dispatch (16x less math than dense reference).
// v2: each wave holds TWO A fragments (all 32 tile rows) and a distinct
//     column quarter -> every B fragment feeds 2 WMMAs (2x L2 intensity).
// ---------------------------------------------------------------------------

typedef __bf16 bf16;
typedef __attribute__((ext_vector_type(16))) __bf16 v16bf;
typedef __attribute__((ext_vector_type(8)))  __bf16 v8bf;
typedef __attribute__((ext_vector_type(8)))  float  v8f;

constexpr int BB   = 2;
constexpr int TT   = 2048;
constexpr int DD   = 5120;   // d_model
constexpr int FDIM = 384;    // ffn dim
constexpr int NE   = 32;     // routed experts
constexpr int NS   = 2;      // shared experts
constexpr int NTOK = BB * TT;      // 4096 tokens
constexpr int HC   = NS * FDIM;    // 768 concatenated shared ffn cols

// ---------------- helpers ----------------
__device__ inline bf16 to_bf16(float f) {
  unsigned u = __builtin_bit_cast(unsigned, f);
  unsigned r = u + 0x7FFFu + ((u >> 16) & 1u);          // round-to-nearest-even
  unsigned short h = (unsigned short)(r >> 16);
  return __builtin_bit_cast(bf16, h);
}

__device__ inline v8f zero8() {
  v8f z;
#pragma unroll
  for (int j = 0; j < 8; ++j) z[j] = 0.0f;
  return z;
}

__device__ inline v16bf cat8(v8bf lo, v8bf hi) {
  return __builtin_shufflevector(lo, hi, 0,1,2,3,4,5,6,7,8,9,10,11,12,13,14,15);
}

// A fragment (16x32, row-major source): lane&15 = row; lane-half K split
// {k..k+7, k+16..k+23} / {k+8..k+15, k+24..k+31}.  p includes row*stride + hi8.
__device__ inline v16bf loadA(const bf16* p) {
  const v8bf lo = *(const v8bf*)p;
  const v8bf hi = *(const v8bf*)(p + 16);
  return cat8(lo, hi);
}

// B fragment (32x16, from pre-transposed [N][K] rows): lane&15 = out col;
// lane-half K split {k..k+7, k+16..k+23} / {k+8..k+15, k+24..k+31}.  p includes col*stride + hi16.
__device__ inline v16bf loadB(const bf16* p) {
  const v8bf lo = *(const v8bf*)p;
  const v8bf hi = *(const v8bf*)(p + 16);
  return cat8(lo, hi);
}

__device__ inline v8f wmma_bf16(v16bf a, v16bf b, v8f c) {
  return __builtin_amdgcn_wmma_f32_16x16x32_bf16(
      /*neg_a=*/false, a, /*neg_b=*/false, b,
      /*c_mod=*/(short)0, c, /*reuse_a=*/false, /*reuse_b=*/false);
}

__device__ inline float gelu_exact(float v) {
  return 0.5f * v * (1.0f + erff(v * 0.70710678118654752440f));
}

__device__ inline float wred_max(float v) {
#pragma unroll
  for (int o = 16; o; o >>= 1) v = fmaxf(v, __shfl_xor(v, o));
  return v;
}
__device__ inline float wred_sum(float v) {
#pragma unroll
  for (int o = 16; o; o >>= 1) v += __shfl_xor(v, o);
  return v;
}
__device__ inline int wred_min(int v) {
#pragma unroll
  for (int o = 16; o; o >>= 1) v = min(v, __shfl_xor(v, o));
  return v;
}

// ---------------- prep kernels ----------------
__global__ void cvt_bf16_kernel(const float* __restrict__ in, bf16* __restrict__ out, size_t n) {
  size_t i = (size_t)blockIdx.x * blockDim.x + threadIdx.x;
  const size_t stride = (size_t)gridDim.x * blockDim.x;
  for (; i < n; i += stride) out[i] = to_bf16(in[i]);
}

// in: [batch][R][C] f32  ->  out: [batch][C][R] bf16
__global__ void transpose_bf16_kernel(const float* __restrict__ in, bf16* __restrict__ out,
                                      int R, int C) {
  const size_t total = (size_t)R * C;
  const size_t base  = (size_t)blockIdx.y * total;
  size_t i = (size_t)blockIdx.x * blockDim.x + threadIdx.x;
  const size_t stride = (size_t)gridDim.x * blockDim.x;
  for (; i < total; i += stride) {
    const int c = (int)(i / R);
    const int r = (int)(i % R);
    out[base + i] = to_bf16(in[base + (size_t)r * C + c]);
  }
}

__global__ void zero_cnt_kernel(int* cnt) {
  if (threadIdx.x < NE) cnt[threadIdx.x] = 0;
}

// ---------------- gating: 1 wave32 per token, lane == expert ----------------
__global__ void __launch_bounds__(128)
moe_gate_kernel(const float* __restrict__ x, const float* __restrict__ gw,
                const float* __restrict__ gb, int* __restrict__ cnt,
                int* __restrict__ atok, float* __restrict__ aw) {
  const int t    = blockIdx.x * 4 + (threadIdx.x >> 5);
  const int lane = threadIdx.x & 31;
  const float* xr = x + (size_t)t * DD;

  float logit = gb[lane];
  for (int d = 0; d < DD; d += 4) {
    const float4 xv = *(const float4*)(xr + d);
    logit = fmaf(xv.x, gw[(size_t)(d + 0) * NE + lane], logit);
    logit = fmaf(xv.y, gw[(size_t)(d + 1) * NE + lane], logit);
    logit = fmaf(xv.z, gw[(size_t)(d + 2) * NE + lane], logit);
    logit = fmaf(xv.w, gw[(size_t)(d + 3) * NE + lane], logit);
  }
  const float mx = wred_max(logit);
  const float ex = expf(logit - mx);
  const float p  = ex / wred_sum(ex);
  // top-2 (ties -> lowest lane, matching lax.top_k); raw probs, no renorm
  const float m1 = wred_max(p);
  const int   l1 = wred_min(p == m1 ? lane : 32);
  const float pm = (lane == l1) ? -1.0f : p;
  const float m2 = wred_max(pm);
  const int   l2 = wred_min(pm == m2 ? lane : 32);

  if (lane == 0) {
    int pos = atomicAdd(&cnt[l1], 1);
    atok[(size_t)l1 * NTOK + pos] = t;
    aw  [(size_t)l1 * NTOK + pos] = m1;
    pos = atomicAdd(&cnt[l2], 1);
    atok[(size_t)l2 * NTOK + pos] = t;
    aw  [(size_t)l2 * NTOK + pos] = m2;
  }
}

// ---------------- shared experts: fused 2-layer FFN, 32-token tile ----------
// Each wave: all 32 rows (2 A frags) x its own column quarter.
// w1t: [NS*FDIM][DD] bf16 (transposed, concatenated); w2t: [NS][DD][FDIM] bf16.
__global__ void __launch_bounds__(128)
moe_shared_ffn_kernel(const bf16* __restrict__ xb, const bf16* __restrict__ w1t,
                      const float* __restrict__ b1, const bf16* __restrict__ w2t,
                      const float* __restrict__ b2, float* __restrict__ out) {
  __shared__ __align__(16) bf16 hs[32 * HC];   // 48 KB of 320 KB WGP LDS

  const int lane = threadIdx.x & 31;
  const int nq   = threadIdx.x >> 5;   // wave id == column quarter
  const int l16  = lane & 15;
  const int hi8  = (lane >> 4) << 3;   // A-frag K offset / C-frag row offset
  const int hi16 = (lane >> 4) << 3;   // B-frag K offset
  const int m0   = blockIdx.x * 32;

  const bf16* aP0 = xb + (size_t)(m0 + l16) * DD + hi8;        // rows 0..15
  const bf16* aP1 = xb + (size_t)(m0 + 16 + l16) * DD + hi8;   // rows 16..31

  // ---- layer 1: h = gelu(x @ W1cat + b1cat); wave cols = nq*192..+191 ----
  for (int nc = 0; nc < 2; ++nc) {
    const int n0 = nq * 192 + nc * 96;
    v8f acc[2][6];
#pragma unroll
    for (int r = 0; r < 2; ++r)
#pragma unroll
      for (int f = 0; f < 6; ++f) acc[r][f] = zero8();

    const bf16* bP[6];
#pragma unroll
    for (int f = 0; f < 6; ++f)
      bP[f] = w1t + (size_t)(n0 + f * 16 + l16) * DD + hi16;

    for (int k = 0; k < DD; k += 32) {
      const v16bf a0 = loadA(aP0 + k);
      const v16bf a1 = loadA(aP1 + k);
#pragma unroll
      for (int f = 0; f < 6; ++f) {
        const v16bf b = loadB(bP[f] + k);        // one B load -> two WMMAs
        acc[0][f] = wmma_bf16(a0, b, acc[0][f]);
        acc[1][f] = wmma_bf16(a1, b, acc[1][f]);
      }
    }
#pragma unroll
    for (int f = 0; f < 6; ++f) {
      const int col = n0 + f * 16 + l16;
      const float bias = b1[col];
#pragma unroll
      for (int r = 0; r < 2; ++r)
#pragma unroll
        for (int j = 0; j < 8; ++j) {
          const int row = r * 16 + j + hi8;
          hs[row * HC + col] = to_bf16(gelu_exact(acc[r][f][j] + bias));
        }
    }
  }
  __syncthreads();

  // ---- layer 2: out = h @ W2cat + sum_s b2_s; wave cols = nq*1280..+1279 ----
  const bf16* aB0 = hs + (size_t)l16 * HC + hi8;
  const bf16* aB1 = hs + (size_t)(16 + l16) * HC + hi8;
  for (int nc = 0; nc < 10; ++nc) {
    const int n0 = nq * 1280 + nc * 128;
    v8f acc[2][8];
#pragma unroll
    for (int r = 0; r < 2; ++r)
#pragma unroll
      for (int f = 0; f < 8; ++f) acc[r][f] = zero8();

    for (int k = 0; k < HC; k += 32) {
      const v16bf a0 = loadA(aB0 + k);
      const v16bf a1 = loadA(aB1 + k);
      const int s  = (k >= FDIM) ? 1 : 0;
      const int kk = k - s * FDIM;
      const bf16* ws = w2t + (size_t)s * DD * FDIM;
#pragma unroll
      for (int f = 0; f < 8; ++f) {
        const v16bf b = loadB(ws + (size_t)(n0 + f * 16 + l16) * FDIM + kk + hi16);
        acc[0][f] = wmma_bf16(a0, b, acc[0][f]);
        acc[1][f] = wmma_bf16(a1, b, acc[1][f]);
      }
    }
#pragma unroll
    for (int f = 0; f < 8; ++f) {
      const int col = n0 + f * 16 + l16;
      const float bias = b2[col] + b2[DD + col];   // shared_b2.sum(0)
#pragma unroll
      for (int r = 0; r < 2; ++r)
#pragma unroll
        for (int j = 0; j < 8; ++j) {
          const int row = m0 + r * 16 + j + hi8;
          out[(size_t)row * DD + col] = acc[r][f][j] + bias;
        }
    }
  }
}

// ---------------- routed experts: gathered token tiles per expert -----------
// w1t: [NE][FDIM][DD] bf16; w2t: [NE][DD][FDIM] bf16.
__global__ void __launch_bounds__(128)
moe_routed_ffn_kernel(const bf16* __restrict__ xb, const bf16* __restrict__ w1t,
                      const float* __restrict__ b1, const bf16* __restrict__ w2t,
                      const float* __restrict__ b2, const int* __restrict__ cnt,
                      const int* __restrict__ atok, const float* __restrict__ aw,
                      float* __restrict__ out) {
  const int e  = blockIdx.x;
  const int m0 = blockIdx.y * 32;
  const int ce = cnt[e];
  if (m0 >= ce) return;

  __shared__ __align__(16) bf16 hs[32 * FDIM];   // 24 KB
  __shared__ int   tok_s[32];
  __shared__ float w_s[32];

  if (threadIdx.x < 32) {
    const int idx = m0 + (int)threadIdx.x;
    const bool v  = idx < ce;
    tok_s[threadIdx.x] = v ? atok[(size_t)e * NTOK + idx] : -1;
    w_s  [threadIdx.x] = v ? aw  [(size_t)e * NTOK + idx] : 0.0f;
  }
  __syncthreads();

  const int lane = threadIdx.x & 31;
  const int nq   = threadIdx.x >> 5;
  const int l16  = lane & 15;
  const int hi8  = (lane >> 4) << 3;
  const int hi16 = (lane >> 4) << 3;

  const int tok0 = tok_s[l16];
  const int tok1 = tok_s[16 + l16];
  const bf16* aP0 = xb + (size_t)(tok0 < 0 ? 0 : tok0) * DD + hi8;  // gathered rows
  const bf16* aP1 = xb + (size_t)(tok1 < 0 ? 0 : tok1) * DD + hi8;

  const bf16*  w1te = w1t + (size_t)e * FDIM * DD;
  const float* b1e  = b1 + (size_t)e * FDIM;

  // ---- layer 1: h = gelu(x_g @ W1_e + b1_e); wave cols = nq*96..+95 ----
  {
    const int n0 = nq * 96;
    v8f acc[2][6];
#pragma unroll
    for (int r = 0; r < 2; ++r)
#pragma unroll
      for (int f = 0; f < 6; ++f) acc[r][f] = zero8();

    const bf16* bP[6];
#pragma unroll
    for (int f = 0; f < 6; ++f)
      bP[f] = w1te + (size_t)(n0 + f * 16 + l16) * DD + hi16;

    for (int k = 0; k < DD; k += 32) {
      const v16bf a0 = loadA(aP0 + k);
      const v16bf a1 = loadA(aP1 + k);
#pragma unroll
      for (int f = 0; f < 6; ++f) {
        const v16bf b = loadB(bP[f] + k);
        acc[0][f] = wmma_bf16(a0, b, acc[0][f]);
        acc[1][f] = wmma_bf16(a1, b, acc[1][f]);
      }
    }
#pragma unroll
    for (int f = 0; f < 6; ++f) {
      const int col = n0 + f * 16 + l16;
      const float bias = b1e[col];
#pragma unroll
      for (int r = 0; r < 2; ++r)
#pragma unroll
        for (int j = 0; j < 8; ++j) {
          const int row = r * 16 + j + hi8;
          hs[row * FDIM + col] = to_bf16(gelu_exact(acc[r][f][j] + bias));
        }
    }
  }
  __syncthreads();

  // ---- layer 2: out[tok] += w * (h @ W2_e + b2_e); token hit exactly 2x ----
  const bf16*  aB0  = hs + (size_t)l16 * FDIM + hi8;
  const bf16*  aB1  = hs + (size_t)(16 + l16) * FDIM + hi8;
  const bf16*  w2te = w2t + (size_t)e * DD * FDIM;
  const float* b2e  = b2 + (size_t)e * DD;

  for (int nc = 0; nc < 10; ++nc) {
    const int n0 = nq * 1280 + nc * 128;
    v8f acc[2][8];
#pragma unroll
    for (int r = 0; r < 2; ++r)
#pragma unroll
      for (int f = 0; f < 8; ++f) acc[r][f] = zero8();

    for (int k = 0; k < FDIM; k += 32) {
      const v16bf a0 = loadA(aB0 + k);
      const v16bf a1 = loadA(aB1 + k);
#pragma unroll
      for (int f = 0; f < 8; ++f) {
        const v16bf b = loadB(w2te + (size_t)(n0 + f * 16 + l16) * FDIM + k + hi16);
        acc[0][f] = wmma_bf16(a0, b, acc[0][f]);
        acc[1][f] = wmma_bf16(a1, b, acc[1][f]);
      }
    }
#pragma unroll
    for (int f = 0; f < 8; ++f) {
      const int col = n0 + f * 16 + l16;
      const float bias = b2e[col];
#pragma unroll
      for (int r = 0; r < 2; ++r)
#pragma unroll
        for (int j = 0; j < 8; ++j) {
          const int rr = r * 16 + j + hi8;
          const int t2 = tok_s[rr];
          if (t2 >= 0)
            unsafeAtomicAdd(out + (size_t)t2 * DD + col, w_s[rr] * (acc[r][f][j] + bias));
        }
    }
  }
}

// ---------------- host launcher ----------------
extern "C" void kernel_launch(void* const* d_in, const int* in_sizes, int n_in,
                              void* d_out, int out_size, void* d_ws, size_t ws_size,
                              hipStream_t stream) {
  (void)in_sizes; (void)n_in; (void)out_size; (void)ws_size;

  const float* x   = (const float*)d_in[0];
  const float* sw1 = (const float*)d_in[1];   // [S, D, FD]
  const float* sb1 = (const float*)d_in[2];   // [S, FD] (contig == [768])
  const float* sw2 = (const float*)d_in[3];   // [S, FD, D]
  const float* sb2 = (const float*)d_in[4];   // [S, D]
  const float* rw1 = (const float*)d_in[5];   // [E, D, FD]
  const float* rb1 = (const float*)d_in[6];   // [E, FD]
  const float* rw2 = (const float*)d_in[7];   // [E, FD, D]
  const float* rb2 = (const float*)d_in[8];   // [E, D]
  const float* gw  = (const float*)d_in[9];   // [D, E]
  const float* gb  = (const float*)d_in[10];  // [E]
  float* out = (float*)d_out;

  // workspace carve (~310 MB total)
  char* ws = (char*)d_ws;
  size_t off = 0;
  auto carve = [&](size_t bytes) -> char* {
    char* p = ws + off;
    off = (off + bytes + 255) & ~(size_t)255;
    return p;
  };
  bf16*  xb   = (bf16*)carve((size_t)NTOK * DD * sizeof(bf16));
  bf16*  w1ts = (bf16*)carve((size_t)NS * FDIM * DD * sizeof(bf16));
  bf16*  w2ts = (bf16*)carve((size_t)NS * DD * FDIM * sizeof(bf16));
  bf16*  w1tr = (bf16*)carve((size_t)NE * FDIM * DD * sizeof(bf16));
  bf16*  w2tr = (bf16*)carve((size_t)NE * DD * FDIM * sizeof(bf16));
  int*   cnt  = (int*)carve(NE * sizeof(int));
  int*   atok = (int*)carve((size_t)NE * NTOK * sizeof(int));
  float* aw   = (float*)carve((size_t)NE * NTOK * sizeof(float));

  // prep: fp32 -> bf16 (x); transpose+convert weights so B fragments are
  // contiguous per lane (W1 -> [*,FD,D], W2 -> [*,D,FD])
  cvt_bf16_kernel<<<2048, 256, 0, stream>>>(x, xb, (size_t)NTOK * DD);
  transpose_bf16_kernel<<<dim3(512, NS), 256, 0, stream>>>(sw1, w1ts, DD, FDIM);
  transpose_bf16_kernel<<<dim3(512, NS), 256, 0, stream>>>(sw2, w2ts, FDIM, DD);
  transpose_bf16_kernel<<<dim3(512, NE), 256, 0, stream>>>(rw1, w1tr, DD, FDIM);
  transpose_bf16_kernel<<<dim3(512, NE), 256, 0, stream>>>(rw2, w2tr, FDIM, DD);

  zero_cnt_kernel<<<1, 64, 0, stream>>>(cnt);
  moe_gate_kernel<<<NTOK / 4, 128, 0, stream>>>(x, gw, gb, cnt, atok, aw);

  // shared FFN writes d_out (overwrites poison); routed FFN atomically adds.
  moe_shared_ffn_kernel<<<NTOK / 32, 128, 0, stream>>>(xb, w1ts, sb1, w2ts, sb2, out);
  moe_routed_ffn_kernel<<<dim3(NE, NTOK / 32), 128, 0, stream>>>(
      xb, w1tr, rb1, w2tr, rb2, cnt, atok, aw, out);
}
